// TransformerBlock_35957466202458
// MI455X (gfx1250) — compile-verified, matrix operand order checked by reading, not run on hardware
//
#include <hip/hip_runtime.h>
#include <hip/hip_bf16.h>

typedef __bf16 bf16;
typedef __attribute__((ext_vector_type(16))) __bf16 v16bf;
typedef __attribute__((ext_vector_type(8)))  __bf16 v8bf;
typedef __attribute__((ext_vector_type(8)))  float  v8f;
typedef __attribute__((__vector_size__(4 * sizeof(int)))) int v4i_vec;

#define D_MODEL 512
#define TCTX    20
#define NHEADS  8
#define HDIM    64
#define HB_LD   (D_MODEL + 8)   // 520 elems = 1040 B (16B multiple)
#define HD_LD   (HDIM + 8)      // 72 elems = 144 B (16B multiple)
#define VT_LD   40              // 40 elems = 80 B (16B multiple)

// ---------------------------------------------------------------------------
// Unified fragment loader. CDNA5 wave32 16-bit A/B layout: per lane,
// row = r0 + (lane&15); K pairs are contiguous, VGPRs 0-3 cover k0+g*8..+7,
// VGPRs 4-7 cover k0+16+g*8..+7 (g = lane>>4). With a K-contiguous source a
// fragment is exactly two 16-byte loads.
__device__ inline v16bf ld_frag(const bf16* base, int ld, int r0, int k0, int lane) {
  int rn = r0 + (lane & 15);
  int g  = lane >> 4;
  const bf16* p = base + (size_t)rn * ld + k0 + g * 8;
  v8bf lo = *(const v8bf*)p;
  v8bf hi = *(const v8bf*)(p + 16);
  v16bf r;
#pragma unroll
  for (int i = 0; i < 8; ++i) { r[i] = lo[i]; r[8 + i] = hi[i]; }
  return r;
}

__device__ inline v8f wmma_bf16(v16bf a, v16bf b, v8f c) {
  return __builtin_amdgcn_wmma_f32_16x16x32_bf16(false, a, false, b, (short)0, c,
                                                 false, false);
}

// C 16x16 f32 layout: lane n = lane&15, row = r + 8*(lane>>4)
__device__ inline void st_c_bf16(bf16* dst, int ld, int row0, int col0, v8f c,
                                 int lane, float scale) {
  int g = lane >> 4, n = lane & 15;
#pragma unroll
  for (int r = 0; r < 8; ++r)
    dst[(row0 + r + 8 * g) * ld + col0 + n] = (bf16)(c[r] * scale);
}

// transposed store (for V -> vtT)
__device__ inline void st_c_bf16_T(bf16* dst, int ld, int row0, int col0, v8f c,
                                   int lane) {
  int g = lane >> 4, n = lane & 15;
#pragma unroll
  for (int r = 0; r < 8; ++r)
    dst[(col0 + n) * ld + row0 + r + 8 * g] = (bf16)c[r];
}

// ---------------- weight conversion pre-pass ----------------
// plain f32 -> bf16
__global__ void cvt_f32_bf16(const float* __restrict__ src, bf16* __restrict__ dst,
                             int n) {
  int i = blockIdx.x * blockDim.x + threadIdx.x;
  if (i < n) dst[i] = (bf16)src[i];
}

// batched transpose+convert: src [Bt][K][N] -> dst [Bt][N][K] (N-major)
__global__ void cvt_t_f32_bf16(const float* __restrict__ src, bf16* __restrict__ dst,
                               int Bt, int K, int N) {
  int i = blockIdx.x * blockDim.x + threadIdx.x;
  int total = Bt * K * N;
  if (i < total) {
    int b = i / (K * N);
    int r = i % (K * N);
    int k = r / N, n = r % N;
    dst[((size_t)b * N + n) * K + k] = (bf16)src[i];
  }
}

// ---------------- fused transformer block: one workgroup per batch ----------------
__global__ __launch_bounds__(256) void transformer_block_kernel(
    const float* __restrict__ x, const float* __restrict__ ln1w,
    const float* __restrict__ ln1b, const float* __restrict__ wo_bias,
    const float* __restrict__ ln2w, const float* __restrict__ ln2b,
    const float* __restrict__ b1, const float* __restrict__ b2,
    const bf16* __restrict__ WqT, const bf16* __restrict__ WkT,
    const bf16* __restrict__ WvT, const bf16* __restrict__ WoT,
    const bf16* __restrict__ W1T, const bf16* __restrict__ W2T,
    float* __restrict__ out) {
  __shared__ float xres[TCTX][D_MODEL];   // residual stream (f32)
  __shared__ bf16  hb[32][HB_LD];         // LN output (bf16, padded to 32 rows)
  __shared__ float sc[32][32];            // raw attention scores
  __shared__ bf16  at[32][32];            // softmax probs (bf16)
  __shared__ bf16  qt[32][HD_LD];         // Q (token-major)
  __shared__ bf16  kt[32][HD_LD];         // K (token-major == N-major for Q.K^T)
  __shared__ bf16  vtT[HDIM][VT_LD];      // V transposed (dim-major)
  __shared__ bf16  ctx[32][HB_LD];        // concat heads
  __shared__ bf16  f1[32][HB_LD];         // FFN hidden chunk (512 cols)

  const int tid  = threadIdx.x;
  const int wave = tid >> 5;
  const int lane = tid & 31;
  const int batch = blockIdx.x;
  const float* xb = x + (size_t)batch * TCTX * D_MODEL;

  // ---- stage x into LDS (gfx1250 async-to-LDS; arg types per probed signature) ----
#if __has_builtin(__builtin_amdgcn_global_load_async_to_lds_b128)
  for (int i = tid * 4; i < TCTX * D_MODEL; i += 256 * 4) {
    __builtin_amdgcn_global_load_async_to_lds_b128(
        (__attribute__((address_space(1))) v4i_vec*)(xb + i),
        (__attribute__((address_space(3))) v4i_vec*)(&xres[0][0] + i), 0, 0);
  }
  asm volatile("s_wait_asynccnt 0x0" ::: "memory");
#else
  for (int i = tid; i < TCTX * D_MODEL; i += 256)
    xres[i / D_MODEL][i % D_MODEL] = xb[i];
#endif
  // zero pad rows of hb once (LN writes only rows < 20)
  for (int i = tid; i < 12 * HB_LD; i += 256)
    hb[20 + i / HB_LD][i % HB_LD] = (bf16)0.0f;
  __syncthreads();

  // ---- LN1: one wave per row ----
  for (int row = wave; row < TCTX; row += 8) {
    float s = 0.f, s2 = 0.f;
#pragma unroll
    for (int j = 0; j < 16; ++j) {
      float v = xres[row][lane + 32 * j];
      s += v; s2 += v * v;
    }
#pragma unroll
    for (int o = 16; o > 0; o >>= 1) {
      s  += __shfl_xor(s,  o, 32);
      s2 += __shfl_xor(s2, o, 32);
    }
    float mu  = s * (1.f / D_MODEL);
    float var = s2 * (1.f / D_MODEL) - mu * mu;
    float inv = rsqrtf(var + 1e-5f);
#pragma unroll
    for (int j = 0; j < 16; ++j) {
      int d = lane + 32 * j;
      hb[row][d] = (bf16)((xres[row][d] - mu) * inv * ln1w[d] + ln1b[d]);
    }
  }
  __syncthreads();

  // ---- attention, head by head ----
  for (int head = 0; head < NHEADS; ++head) {
    if (head + 1 < NHEADS && lane == 0)
      __builtin_prefetch(WqT + (size_t)(head + 1) * HDIM * D_MODEL, 0, 0);

    // Q/K/V projections: 24 16x16 tiles (3 mats * 2 mt * 4 nt), 3 per wave.
    // Weights are N-major: W*T[h][e][d], so B fragments are K-contiguous.
    for (int t = wave; t < 24; t += 8) {
      int mat = t >> 3, idx = t & 7;
      int mt = idx >> 2, nt = idx & 3;
      const bf16* W =
          (mat == 0 ? WqT : mat == 1 ? WkT : WvT) + (size_t)head * HDIM * D_MODEL;
      v8f c = {};
      for (int k0 = 0; k0 < D_MODEL; k0 += 32) {
        v16bf a = ld_frag(&hb[0][0], HB_LD, mt * 16, k0, lane);
        v16bf b = ld_frag(W, D_MODEL, nt * 16, k0, lane);
        c = wmma_bf16(a, b, c);
      }
      if (mat == 0)      st_c_bf16(&qt[0][0], HD_LD, mt * 16, nt * 16, c, lane, 0.125f);
      else if (mat == 1) st_c_bf16(&kt[0][0], HD_LD, mt * 16, nt * 16, c, lane, 1.0f);
      else               st_c_bf16_T(&vtT[0][0], VT_LD, mt * 16, nt * 16, c, lane);
    }
    __syncthreads();

    // scores = (Q/8) . K^T : kt is token-major == N-major here
    if (wave < 4) {
      int mt = wave >> 1, nt = wave & 1;
      v8f c = {};
#pragma unroll
      for (int k0 = 0; k0 < HDIM; k0 += 32) {
        v16bf a = ld_frag(&qt[0][0], HD_LD, mt * 16, k0, lane);
        v16bf b = ld_frag(&kt[0][0], HD_LD, nt * 16, k0, lane);
        c = wmma_bf16(a, b, c);
      }
      int g = lane >> 4, n = lane & 15;
#pragma unroll
      for (int r = 0; r < 8; ++r)
        sc[mt * 16 + r + 8 * g][nt * 16 + n] = c[r];
    }
    __syncthreads();

    // causal softmax: one wave per row (lane = column)
    for (int row = wave; row < 32; row += 8) {
      if (row < TCTX) {
        float v = (lane <= row) ? sc[row][lane] : -3.0e38f;
        float m = v;
#pragma unroll
        for (int o = 16; o > 0; o >>= 1) m = fmaxf(m, __shfl_xor(m, o, 32));
        float p = __expf(v - m);
        float s = p;
#pragma unroll
        for (int o = 16; o > 0; o >>= 1) s += __shfl_xor(s, o, 32);
        at[row][lane] = (bf16)(p / s);
      } else {
        at[row][lane] = (bf16)0.0f;
      }
    }
    __syncthreads();

    // ctx_head = attn . V : vtT is dim-major -> K-contiguous B fragments
    {
      int mt = wave >> 2, nt = wave & 3;
      v16bf a = ld_frag(&at[0][0], 32, mt * 16, 0, lane);
      v16bf b = ld_frag(&vtT[0][0], VT_LD, nt * 16, 0, lane);
      v8f c = {};
      c = wmma_bf16(a, b, c);
      st_c_bf16(&ctx[0][0], HB_LD, mt * 16, head * HDIM + nt * 16, c, lane, 1.0f);
    }
    __syncthreads();
  }

  // ---- attn_out = ctx . Wo^T + bias; residual add into xres ----
  // Wave owns mt = wave>>2, N-tiles (wave&3)*8 .. +7: hoist A fragment per k-step.
  {
    int mt  = wave >> 2;
    int ntb = (wave & 3) * 8;
    v8f c8[8];
#pragma unroll
    for (int i = 0; i < 8; ++i) c8[i] = (v8f){};
    for (int k0 = 0; k0 < D_MODEL; k0 += 32) {
      v16bf a = ld_frag(&ctx[0][0], HB_LD, mt * 16, k0, lane);
#pragma unroll
      for (int i = 0; i < 8; ++i) {
        v16bf b = ld_frag(WoT, D_MODEL, (ntb + i) * 16, k0, lane);
        c8[i] = wmma_bf16(a, b, c8[i]);
      }
    }
    int g = lane >> 4;
#pragma unroll
    for (int i = 0; i < 8; ++i) {
      int n0 = (ntb + i) * 16 + (lane & 15);
      float bias = wo_bias[n0];
#pragma unroll
      for (int r = 0; r < 8; ++r) {
        int row = mt * 16 + r + 8 * g;
        if (row < TCTX) xres[row][n0] += c8[i][r] + bias;
      }
    }
  }
  __syncthreads();

  // ---- LN2 into hb ----
  for (int row = wave; row < TCTX; row += 8) {
    float s = 0.f, s2 = 0.f;
#pragma unroll
    for (int j = 0; j < 16; ++j) {
      float v = xres[row][lane + 32 * j];
      s += v; s2 += v * v;
    }
#pragma unroll
    for (int o = 16; o > 0; o >>= 1) {
      s  += __shfl_xor(s,  o, 32);
      s2 += __shfl_xor(s2, o, 32);
    }
    float mu  = s * (1.f / D_MODEL);
    float var = s2 * (1.f / D_MODEL) - mu * mu;
    float inv = rsqrtf(var + 1e-5f);
#pragma unroll
    for (int j = 0; j < 16; ++j) {
      int d = lane + 32 * j;
      hb[row][d] = (bf16)((xres[row][d] - mu) * inv * ln2w[d] + ln2b[d]);
    }
  }
  __syncthreads();

  // ---- FFN: hidden dim 2048 chunked 4 x 512; ff2 accumulates in registers ----
  const int mt  = wave >> 2;
  const int ntb = (wave & 3) * 8;
  v8f acc[8];
#pragma unroll
  for (int i = 0; i < 8; ++i) acc[i] = (v8f){};

  for (int ch = 0; ch < 4; ++ch) {
    // f1 = relu(hb . W1[:, ch*512 .. +512] + b1); W1T is [2048][512] N-major
    {
      v8f c8[8];
#pragma unroll
      for (int i = 0; i < 8; ++i) c8[i] = (v8f){};
      for (int k0 = 0; k0 < D_MODEL; k0 += 32) {
        v16bf a = ld_frag(&hb[0][0], HB_LD, mt * 16, k0, lane);
#pragma unroll
        for (int i = 0; i < 8; ++i) {
          v16bf b = ld_frag(W1T, D_MODEL, ch * 512 + (ntb + i) * 16, k0, lane);
          c8[i] = wmma_bf16(a, b, c8[i]);
        }
      }
      int g = lane >> 4, n = lane & 15;
#pragma unroll
      for (int i = 0; i < 8; ++i) {
        float bias = b1[ch * 512 + (ntb + i) * 16 + n];
#pragma unroll
        for (int r = 0; r < 8; ++r) {
          float v = c8[i][r] + bias;
          f1[mt * 16 + r + 8 * g][(ntb + i) * 16 + n] = (bf16)(v > 0.f ? v : 0.f);
        }
      }
    }
    __syncthreads();

    // ff2 += f1 . W2[ch*512 .. +512, :]; W2T is [512][2048] N-major
    for (int k0 = 0; k0 < 512; k0 += 32) {
      v16bf a = ld_frag(&f1[0][0], HB_LD, mt * 16, k0, lane);
#pragma unroll
      for (int i = 0; i < 8; ++i) {
        v16bf b = ld_frag(W2T, 4 * D_MODEL, (ntb + i) * 16, ch * 512 + k0, lane);
        acc[i] = wmma_bf16(a, b, acc[i]);
      }
    }
    __syncthreads();
  }

  // ---- epilogue: out = xres + ff2 + b2 ----
  float* ob = out + (size_t)batch * TCTX * D_MODEL;
  int g = lane >> 4;
#pragma unroll
  for (int i = 0; i < 8; ++i) {
    int n0 = (ntb + i) * 16 + (lane & 15);
    float bias = b2[n0];
#pragma unroll
    for (int r = 0; r < 8; ++r) {
      int row = mt * 16 + r + 8 * g;
      if (row < TCTX)
        ob[(size_t)row * D_MODEL + n0] = xres[row][n0] + acc[i][r] + bias;
    }
  }
}

// ---------------- host launcher ----------------
extern "C" void kernel_launch(void* const* d_in, const int* in_sizes, int n_in,
                              void* d_out, int out_size, void* d_ws, size_t ws_size,
                              hipStream_t stream) {
  const float* x    = (const float*)d_in[0];
  const float* ln1w = (const float*)d_in[1];
  const float* ln1b = (const float*)d_in[2];
  const float* Wq   = (const float*)d_in[3];
  const float* Wk   = (const float*)d_in[4];
  const float* Wv   = (const float*)d_in[5];
  const float* Wo   = (const float*)d_in[6];
  const float* Wob_ = (const float*)d_in[7];
  const float* ln2w = (const float*)d_in[8];
  const float* ln2b = (const float*)d_in[9];
  const float* W1   = (const float*)d_in[10];
  const float* b1   = (const float*)d_in[11];
  const float* W2   = (const float*)d_in[12];
  const float* b2   = (const float*)d_in[13];
  float* out = (float*)d_out;

  const int B = in_sizes[0] / (TCTX * D_MODEL);

  // bf16 weight workspace (element offsets); all stored N-major for the GEMMs
  bf16* ws  = (bf16*)d_ws;
  bf16* WqT = ws;            // [8][64][512]  = 262144
  bf16* WkT = ws + 262144;   // 262144
  bf16* WvT = ws + 524288;   // 262144
  bf16* WoT = ws + 786432;   // Wo rows are already the N dim (ctx @ Wo.T)
  bf16* W1T = ws + 1048576;  // [2048][512]   = 1048576
  bf16* W2T = ws + 2097152;  // [512][2048]   = 1048576

  auto cvtT = [&](const float* s, bf16* d, int Bt, int K, int N) {
    int n = Bt * K * N;
    cvt_t_f32_bf16<<<(n + 255) / 256, 256, 0, stream>>>(s, d, Bt, K, N);
  };
  cvtT(Wq, WqT, NHEADS, D_MODEL, HDIM);
  cvtT(Wk, WkT, NHEADS, D_MODEL, HDIM);
  cvtT(Wv, WvT, NHEADS, D_MODEL, HDIM);
  cvt_f32_bf16<<<(262144 + 255) / 256, 256, 0, stream>>>(Wo, WoT, 262144);
  cvtT(W1, W1T, 1, D_MODEL, 4 * D_MODEL);
  cvtT(W2, W2T, 1, 4 * D_MODEL, D_MODEL);

  transformer_block_kernel<<<B, 256, 0, stream>>>(
      x, ln1w, ln1b, Wob_, ln2w, ln2b, b1, b2,
      WqT, WkT, WvT, WoT, W1T, W2T, out);
}
